// EfficientGraphFusion_22763326669308
// MI455X (gfx1250) — matrix operand order repaired and checked by reading, not compiled, not run on hardware
//
#include <hip/hip_runtime.h>
#include <hip/hip_bf16.h>
#include <math.h>

#define NC    2048
#define HIDN  256
#define NH    4
#define DH_   64
#define SEQL  8
#define KCH   4            // K-split chunks for the attention GEMM
#define KLEN  (NC/KCH)     // 512 columns per chunk

typedef __attribute__((ext_vector_type(16))) __bf16 v16bf;
typedef __attribute__((ext_vector_type(8)))  float  v8f;
typedef __attribute__((ext_vector_type(2)))  float  v2f;

__device__ __forceinline__ float sigm_(float x){ return 1.f/(1.f+__expf(-x)); }

// ---------------------------------------------------------------------------
// graph-builder folded vectors: va = W1a^T w2, vb = W1b^T w2, c = b1.w2 + b2
__global__ void k_gbvec(const float* __restrict__ gbW1, const float* __restrict__ gbb1,
                        const float* __restrict__ gbw2, const float* __restrict__ gbb2,
                        float* __restrict__ va, float* __restrict__ vb, float* __restrict__ cbuf){
  int d = threadIdx.x;                       // 0..255
  float sa = 0.f, sb = 0.f;
  for (int k = 0; k < 128; ++k){
    float w = gbw2[k];
    sa += gbW1[k*(2*HIDN) + d]        * w;
    sb += gbW1[k*(2*HIDN) + HIDN + d] * w;
  }
  va[d] = sa; vb[d] = sb;
  if (d == 0){
    float c = gbb2[0];
    for (int k = 0; k < 128; ++k) c += gbb1[k]*gbw2[k];
    cbuf[0] = c;
  }
}

// ---------------------------------------------------------------------------
// layernorm(embed) -> emb, plus per-node u = emb.va, v = emb.vb
__global__ void k_ln_uv(const float* __restrict__ embed, const float* __restrict__ g,
                        const float* __restrict__ b, const float* __restrict__ va,
                        const float* __restrict__ vb, float* __restrict__ emb,
                        float* __restrict__ u, float* __restrict__ v){
  __shared__ float red[HIDN];
  int row = blockIdx.x, d = threadIdx.x;
  float x = embed[row*HIDN + d];
  red[d] = x; __syncthreads();
  for (int s = 128; s > 0; s >>= 1){ if (d < s) red[d] += red[d+s]; __syncthreads(); }
  float mean = red[0] * (1.f/HIDN); __syncthreads();
  float xm = x - mean;
  red[d] = xm*xm; __syncthreads();
  for (int s = 128; s > 0; s >>= 1){ if (d < s) red[d] += red[d+s]; __syncthreads(); }
  float var = red[0] * (1.f/HIDN); __syncthreads();
  float y = xm * rsqrtf(var + 1e-5f) * g[d] + b[d];
  emb[row*HIDN + d] = y;
  red[d] = y * va[d]; __syncthreads();
  for (int s = 128; s > 0; s >>= 1){ if (d < s) red[d] += red[d+s]; __syncthreads(); }
  if (d == 0) u[row] = red[0];
  __syncthreads();
  red[d] = y * vb[d]; __syncthreads();
  for (int s = 128; s > 0; s >>= 1){ if (d < s) red[d] += red[d+s]; __syncthreads(); }
  if (d == 0) v[row] = red[0];
}

// ---------------------------------------------------------------------------
// fused adjacency positivity -> bitmask (1 u32 per 32 columns)
__global__ void k_adjbits(const float* __restrict__ adj, const float* __restrict__ u,
                          const float* __restrict__ v, const float* __restrict__ cbuf,
                          const float* __restrict__ alp, const float* __restrict__ spw,
                          const float* __restrict__ spb, unsigned* __restrict__ adjb){
  int t = blockIdx.x*blockDim.x + threadIdx.x;   // word index: row*64 + w
  int row = t >> 6, w = t & 63;
  float alpha = alp[0], sw = spw[0], sb = spb[0], c = cbuf[0];
  float vi = v[row];
  unsigned bits = 0u;
  for (int bb = 0; bb < 32; ++bb){
    int j = w*32 + bb;
    float pa = adj[(size_t)row*NC + j]*sw + sb;
    float dyn = 0.f;
    if (pa > 0.f) dyn = sigm_(u[j] + vi + c);     // dyn[i,j] = Wpair[j,i] on symmetric mask
    float fused = pa*(1.f - alpha) + dyn*alpha;
    if (fused > 0.f) bits |= (1u << bb);
  }
  adjb[t] = bits;
}

// ---------------------------------------------------------------------------
// h = emb @ W_gat  (fp32 WMMA 16x16x4).  One wave = 16 rows x one head (64 cols).
// Also emits hT (bf16, [col][node]) and a_src/a_dst per head.
__global__ void k_hgemm(const float* __restrict__ emb, const float* __restrict__ Wgat,
                        const float* __restrict__ attnS, const float* __restrict__ attnD,
                        __bf16* __restrict__ hT, float* __restrict__ a_src,
                        float* __restrict__ a_dst){
  int wid = threadIdx.x >> 5, lane = threadIdx.x & 31;
  int gw = blockIdx.x*8 + wid;
  int mtile = gw >> 2, hd = gw & 3;
  int m0 = mtile*16, lrow = lane & 15;
  int arow = m0 + lrow;
  int koff = (lane < 16) ? 0 : 2;
  v8f acc[4] = {};
  for (int k0 = 0; k0 < HIDN; k0 += 4){
    float2 av = *(const float2*)(emb + (size_t)arow*HIDN + k0 + koff);
    v2f a; a[0] = av.x; a[1] = av.y;
#pragma unroll
    for (int t = 0; t < 4; ++t){
      int cl = t*16 + lrow;                                // local col inside head
      const float* bp = Wgat + ((size_t)hd*HIDN + (k0 + koff))*DH_ + cl;
      v2f b; b[0] = bp[0]; b[1] = bp[DH_];
      acc[t] = __builtin_amdgcn_wmma_f32_16x16x4_f32(false, a, false, b,
                                                     (short)0, acc[t], false, false);
    }
  }
  float sv[4], dv[4];
#pragma unroll
  for (int t = 0; t < 4; ++t){
    sv[t] = attnS[hd*DH_ + t*16 + lrow];
    dv[t] = attnD[hd*DH_ + t*16 + lrow];
  }
#pragma unroll
  for (int r = 0; r < 8; ++r){
    int row = m0 + r + ((lane < 16) ? 0 : 8);
    float ps = 0.f, pd = 0.f;
#pragma unroll
    for (int t = 0; t < 4; ++t){
      float cv = acc[t][r];
      ps += cv*sv[t]; pd += cv*dv[t];
      hT[(size_t)(hd*DH_ + t*16 + lrow)*NC + row] = (__bf16)cv;
    }
    for (int off = 1; off < 16; off <<= 1){
      ps += __shfl_xor(ps, off, 32);
      pd += __shfl_xor(pd, off, 32);
    }
    if (lrow == 0){ a_src[hd*NC + row] = ps; a_dst[hd*NC + row] = pd; }
  }
}

// ---------------------------------------------------------------------------
// per-step: pack active bitmask + count
__global__ void k_stepinit(const int* __restrict__ ts_t, unsigned* __restrict__ actb,
                           int* __restrict__ nact){
  __shared__ int cnt[256];
  int tid = threadIdx.x;
  if (tid < 64){
    unsigned bits = 0u;
    for (int i = 0; i < 32; ++i) if (ts_t[tid*32 + i] > 0) bits |= (1u << i);
    actb[tid] = bits;
  }
  int c = 0;
  for (int it = 0; it < 8; ++it) c += (ts_t[it*256 + tid] > 0) ? 1 : 0;
  cnt[tid] = c; __syncthreads();
  for (int s = 128; s > 0; s >>= 1){ if (tid < s) cnt[tid] += cnt[tid+s]; __syncthreads(); }
  if (tid == 0) nact[0] = cnt[0];
}

// ---------------------------------------------------------------------------
// per (head,row): rowmax over active cols, then sum of exp -> stats (m, 1/(sum+1e-8))
__global__ void k_rowstats(const float* __restrict__ a_src, const float* __restrict__ a_dst,
                           const unsigned* __restrict__ adjb, const int* __restrict__ ts_t,
                           float2* __restrict__ stats){
  __shared__ float red[256];
  int hd = blockIdx.x >> 11, row = blockIdx.x & (NC-1);
  int tid = threadIdx.x;
  bool act_i = ts_t[row] > 0;
  float di = a_dst[hd*NC + row];
  const float* asrc = a_src + hd*NC;
  float m = -__builtin_inff();
  for (int it = 0; it < 8; ++it){
    int j = it*256 + tid;
    if (ts_t[j] > 0){
      float sm = 0.f;
      if (act_i && ((adjb[row*64 + (j>>5)] >> (j&31)) & 1u)){
        float r0 = di + asrc[j];
        sm = (r0 > 0.f) ? r0 : 0.2f*r0;
      }
      m = fmaxf(m, sm);
    }
  }
  red[tid] = m; __syncthreads();
  for (int s = 128; s > 0; s >>= 1){ if (tid < s) red[tid] = fmaxf(red[tid], red[tid+s]); __syncthreads(); }
  m = red[0]; __syncthreads();
  if (!(m > -1e30f)) m = 0.f;                    // no active columns -> rowmax := 0
  float ssum = 0.f;
  for (int it = 0; it < 8; ++it){
    int j = it*256 + tid;
    if (act_i && ts_t[j] > 0 && ((adjb[row*64 + (j>>5)] >> (j&31)) & 1u)){
      float r0 = di + asrc[j];
      float sm = (r0 > 0.f) ? r0 : 0.2f*r0;
      ssum += __expf(sm - m);
    }
  }
  red[tid] = ssum; __syncthreads();
  for (int s = 128; s > 0; s >>= 1){ if (tid < s) red[tid] += red[tid+s]; __syncthreads(); }
  if (tid == 0) stats[hd*NC + row] = make_float2(m, 1.f/(red[0] + 1e-8f));
}

// ---------------------------------------------------------------------------
// fused masked-softmax x V via bf16 WMMA 16x16x32, K-split for occupancy.
// grid (NC/16, KCH), block 128 = 4 waves = 4 heads.  Each wave: 16 rows x
// 64 cols (one head), K = KLEN columns of one chunk.  Writes fp32 partials.
__global__ void k_attn_part(const unsigned* __restrict__ actb, const float* __restrict__ a_srcv,
                            const float* __restrict__ a_dstv, const float2* __restrict__ stats,
                            const unsigned* __restrict__ adjb, const __bf16* __restrict__ hT,
                            float* __restrict__ part){
  int hd = threadIdx.x >> 5, lane = threadIdx.x & 31;
  int m0 = blockIdx.x*16, lrow = lane & 15;
  int chunk = blockIdx.y;
  int arow = m0 + lrow;
  bool act_i = (actb[arow >> 5] >> (arow & 31)) & 1u;
  float di = a_dstv[hd*NC + arow];
  float2 st = stats[hd*NC + arow];
  float mi = st.x, inv = st.y;
  const float* asrc = a_srcv + hd*NC;
  int ao1 = (lane < 16) ? 0 : 16;    // A-fragment K bases (16 contiguous K per half, matches B)
  int ao2 = (lane < 16) ? 8 : 24;
  int bo  = (lane < 16) ? 0 : 16;    // B-fragment K base (16 contiguous K per half)
  v8f acc[4] = {};
  int jlo = chunk*KLEN, jhi = jlo + KLEN;
  for (int j0 = jlo; j0 < jhi; j0 += 32){
    unsigned msk = act_i ? (adjb[arow*64 + (j0>>5)] & actb[j0>>5]) : 0u;
    int b1 = j0 + ao1, b2 = j0 + ao2;
    v16bf A;
#pragma unroll
    for (int s = 0; s < 8; ++s){
      float w1 = 0.f, w2 = 0.f;
      if ((msk >> ((b1 + s) & 31)) & 1u){
        float r0 = di + asrc[b1 + s];
        float sm = (r0 > 0.f) ? r0 : 0.2f*r0;
        w1 = __expf(sm - mi) * inv;
      }
      if ((msk >> ((b2 + s) & 31)) & 1u){
        float r0 = di + asrc[b2 + s];
        float sm = (r0 > 0.f) ? r0 : 0.2f*r0;
        w2 = __expf(sm - mi) * inv;
      }
      A[s]     = (__bf16)w1;
      A[8 + s] = (__bf16)w2;
    }
#pragma unroll
    for (int t = 0; t < 4; ++t){
      const __bf16* hrow = hT + (size_t)(hd*DH_ + t*16 + lrow)*NC;
      v16bf B = *(const v16bf*)(hrow + j0 + bo);           // 32B contiguous
      acc[t] = __builtin_amdgcn_wmma_f32_16x16x32_bf16(false, A, false, B,
                                                       (short)0, acc[t], false, false);
    }
  }
  float* pbase = part + (size_t)chunk*NC*HIDN;
#pragma unroll
  for (int r = 0; r < 8; ++r){
    int row = m0 + r + ((lane < 16) ? 0 : 8);
#pragma unroll
    for (int t = 0; t < 4; ++t){
      pbase[(size_t)row*HIDN + hd*DH_ + t*16 + lrow] = acc[t][r];
    }
  }
}

// sum K-chunks deterministically, apply elu -> merged
__global__ void k_attn_red(const float* __restrict__ part, float* __restrict__ merged){
  int row = blockIdx.x, d = threadIdx.x;
  float s = 0.f;
#pragma unroll
  for (int c = 0; c < KCH; ++c) s += part[((size_t)c*NC + row)*HIDN + d];
  merged[(size_t)row*HIDN + d] = (s > 0.f) ? s : (__expf(s) - 1.f);
}

// ---------------------------------------------------------------------------
// y = merged @ Wm^T + bm   (fp32 WMMA 16x16x4)
__global__ void k_gemm2(const float* __restrict__ merged, const float* __restrict__ Wm,
                        const float* __restrict__ bm, float* __restrict__ ybuf){
  int wid = threadIdx.x >> 5, lane = threadIdx.x & 31;
  int gw = blockIdx.x*8 + wid;
  int mtile = gw >> 4, ntile = gw & 15;
  int m0 = mtile*16, n0 = ntile*16, lrow = lane & 15;
  int arow = m0 + lrow, bcol = n0 + lrow;
  int koff = (lane < 16) ? 0 : 2;
  v8f c = {};
  for (int k0 = 0; k0 < HIDN; k0 += 4){
    float2 av = *(const float2*)(merged + (size_t)arow*HIDN + k0 + koff);
    float2 bv = *(const float2*)(Wm + (size_t)bcol*HIDN + k0 + koff);   // B[k][n]=Wm[n][k]
    v2f a; a[0] = av.x; a[1] = av.y;
    v2f b; b[0] = bv.x; b[1] = bv.y;
    c = __builtin_amdgcn_wmma_f32_16x16x4_f32(false, a, false, b, (short)0, c, false, false);
  }
  float bias = bm[bcol];
#pragma unroll
  for (int r = 0; r < 8; ++r){
    int row = m0 + r + ((lane < 16) ? 0 : 8);
    ybuf[(size_t)row*HIDN + bcol] = c[r] + bias;
  }
}

// ---------------------------------------------------------------------------
// layernorm rows of y, masked by active -> aggm (0 rows for inactive nodes)
__global__ void k_lnred(const float* __restrict__ ybuf, const float* __restrict__ g,
                        const float* __restrict__ b, const int* __restrict__ ts_t,
                        float* __restrict__ aggm){
  __shared__ float red[HIDN];
  int row = blockIdx.x, d = threadIdx.x;
  float x = ybuf[(size_t)row*HIDN + d];
  red[d] = x; __syncthreads();
  for (int s = 128; s > 0; s >>= 1){ if (d < s) red[d] += red[d+s]; __syncthreads(); }
  float mean = red[0]*(1.f/HIDN); __syncthreads();
  float xm = x - mean;
  red[d] = xm*xm; __syncthreads();
  for (int s = 128; s > 0; s >>= 1){ if (d < s) red[d] += red[d+s]; __syncthreads(); }
  float var = red[0]*(1.f/HIDN);
  float y = xm * rsqrtf(var + 1e-5f) * g[d] + b[d];
  aggm[(size_t)row*HIDN + d] = (ts_t[row] > 0) ? y : 0.f;
}

// deterministic column sum over 2048 rows
__global__ void k_colsum(const float* __restrict__ aggm, float* __restrict__ repacc){
  __shared__ float red[256];
  int d = blockIdx.x, tid = threadIdx.x;
  float s = 0.f;
  for (int i = tid; i < NC; i += 256) s += aggm[(size_t)i*HIDN + d];
  red[tid] = s; __syncthreads();
  for (int st = 128; st > 0; st >>= 1){ if (tid < st) red[tid] += red[tid+st]; __syncthreads(); }
  if (tid == 0) repacc[d] = red[0];
}

__global__ void k_finalize(const float* __restrict__ repacc, const int* __restrict__ nact,
                           const float* __restrict__ prev, float* __restrict__ rep_out){
  int d = threadIdx.x;
  int n = nact[0];
  rep_out[d] = (n > 0) ? (repacc[d] / fmaxf((float)n, 1.f)) : prev[d];
}

// ---------------------------------------------------------------------------
// GRU over SEQ reps (single block; tiny)
__global__ void k_gru(const float* __restrict__ reps, const float* __restrict__ wih,
                      const float* __restrict__ whh, const float* __restrict__ bih,
                      const float* __restrict__ bhh, float* __restrict__ out){
  __shared__ float h[HIDN], x[HIDN];
  int d = threadIdx.x;
  h[d] = 0.f; __syncthreads();
  for (int t = 0; t < SEQL; ++t){
    x[d] = reps[t*HIDN + d]; __syncthreads();
    float gi0 = bih[d], gi1 = bih[HIDN + d], gi2 = bih[2*HIDN + d];
    float gh0 = bhh[d], gh1 = bhh[HIDN + d], gh2 = bhh[2*HIDN + d];
    for (int k = 0; k < HIDN; ++k){
      float xv = x[k], hv = h[k];
      gi0 += wih[(size_t)d*HIDN + k]*xv;
      gi1 += wih[(size_t)(HIDN + d)*HIDN + k]*xv;
      gi2 += wih[(size_t)(2*HIDN + d)*HIDN + k]*xv;
      gh0 += whh[(size_t)d*HIDN + k]*hv;
      gh1 += whh[(size_t)(HIDN + d)*HIDN + k]*hv;
      gh2 += whh[(size_t)(2*HIDN + d)*HIDN + k]*hv;
    }
    float r = sigm_(gi0 + gh0);
    float z = sigm_(gi1 + gh1);
    float n = tanhf(gi2 + r*gh2);
    float hn = (1.f - z)*n + z*h[d];
    __syncthreads();
    h[d] = hn; __syncthreads();
  }
  out[d] = h[d];
}

// ---------------------------------------------------------------------------
extern "C" void kernel_launch(void* const* d_in, const int* in_sizes, int n_in,
                              void* d_out, int out_size, void* d_ws, size_t ws_size,
                              hipStream_t stream){
  (void)in_sizes; (void)n_in; (void)out_size; (void)ws_size;
  const int*   ts     = (const int*)  d_in[0];
  const float* adj    = (const float*)d_in[1];
  const float* embed  = (const float*)d_in[2];
  const float* ln1g   = (const float*)d_in[3];
  const float* ln1b   = (const float*)d_in[4];
  const float* Wgat   = (const float*)d_in[5];
  const float* attnS  = (const float*)d_in[6];
  const float* attnD  = (const float*)d_in[7];
  const float* Wm     = (const float*)d_in[8];
  const float* bm     = (const float*)d_in[9];
  const float* ln2g   = (const float*)d_in[10];
  const float* ln2b   = (const float*)d_in[11];
  const float* gbW1   = (const float*)d_in[12];
  const float* gbb1   = (const float*)d_in[13];
  const float* gbw2   = (const float*)d_in[14];
  const float* gbb2   = (const float*)d_in[15];
  const float* alpha  = (const float*)d_in[16];
  const float* wih    = (const float*)d_in[17];
  const float* whh    = (const float*)d_in[18];
  const float* bih    = (const float*)d_in[19];
  const float* bhh    = (const float*)d_in[20];
  const float* nocode = (const float*)d_in[21];
  const float* spw    = (const float*)d_in[22];
  const float* spb    = (const float*)d_in[23];

  char* base = (char*)d_ws;
  size_t off = 0;
  auto take = [&](size_t bytes)->char*{
    char* p = base + off;
    off = (off + bytes + 255) & ~(size_t)255;
    return p;
  };
  float*    emb    = (float*)   take((size_t)NC*HIDN*4);
  float*    va     = (float*)   take(HIDN*4);
  float*    vb     = (float*)   take(HIDN*4);
  float*    cbuf   = (float*)   take(4);
  float*    u      = (float*)   take(NC*4);
  float*    v      = (float*)   take(NC*4);
  unsigned* adjb   = (unsigned*)take((size_t)NC*64*4);
  __bf16*   hT     = (__bf16*)  take((size_t)HIDN*NC*2);
  float*    a_src  = (float*)   take((size_t)NH*NC*4);
  float*    a_dst  = (float*)   take((size_t)NH*NC*4);
  float2*   stats  = (float2*)  take((size_t)NH*NC*8);
  unsigned* actb   = (unsigned*)take(64*4);
  int*      nact   = (int*)     take(4);
  float*    part   = (float*)   take((size_t)KCH*NC*HIDN*4);
  float*    merged = (float*)   take((size_t)NC*HIDN*4);
  float*    ybuf   = (float*)   take((size_t)NC*HIDN*4);
  float*    repacc = (float*)   take(HIDN*4);
  float*    reps   = (float*)   take((size_t)SEQL*HIDN*4);

  // static (timestep-independent) precompute
  k_gbvec  <<<1, 256, 0, stream>>>(gbW1, gbb1, gbw2, gbb2, va, vb, cbuf);
  k_ln_uv  <<<NC, 256, 0, stream>>>(embed, ln1g, ln1b, va, vb, emb, u, v);
  k_adjbits<<<NC*64/256, 256, 0, stream>>>(adj, u, v, cbuf, alpha, spw, spb, adjb);
  k_hgemm  <<<(NC/16)*NH/8, 256, 0, stream>>>(emb, Wgat, attnS, attnD, hT, a_src, a_dst);

  for (int t = 0; t < SEQL; ++t){
    const int* ts_t = ts + t*NC;
    k_stepinit <<<1, 256, 0, stream>>>(ts_t, actb, nact);
    k_rowstats <<<NH*NC, 256, 0, stream>>>(a_src, a_dst, adjb, ts_t, stats);
    k_attn_part<<<dim3(NC/16, KCH), 128, 0, stream>>>(actb, a_src, a_dst, stats, adjb, hT, part);
    k_attn_red <<<NC, 256, 0, stream>>>(part, merged);
    k_gemm2    <<<(NC/16)*(HIDN/16)/8, 256, 0, stream>>>(merged, Wm, bm, ybuf);
    k_lnred    <<<NC, 256, 0, stream>>>(ybuf, ln2g, ln2b, ts_t, merged);   // reuse merged as aggm
    k_colsum   <<<HIDN, 256, 0, stream>>>(merged, repacc);
    const float* prev = (t == 0) ? nocode : (const float*)(reps + (t-1)*HIDN);
    k_finalize <<<1, HIDN, 0, stream>>>(repacc, nact, prev, reps + t*HIDN);
  }

  k_gru<<<1, HIDN, 0, stream>>>(reps, wih, whh, bih, bhh, (float*)d_out);
}
